// SimpleMetadataEncoder_87617332838624
// MI455X (gfx1250) — matrix operand order repaired and checked by reading, not compiled, not run on hardware
//
#include <hip/hip_runtime.h>
#include <hip/hip_bf16.h>
#include <stdint.h>

#define ARTIST_V 16384
#define GENRE_V  2048
#define TOTAL    18433   // ARTIST_V + GENRE_V + 1
#define HIDDEN   9216
#define OUT_DIM  64
#define BATCH    4096
#define JT       16      // j-tile per workgroup in phase 1
#define ROWPITCH 18440   // padded row length (floats) for double buffer
#define ASYNC_PER_ROW 19 // uniform async issues per thread per row (19*1024 >= 18433)

typedef _Float16 v16h __attribute__((ext_vector_type(16)));
typedef _Float16 v8h  __attribute__((ext_vector_type(8)));
typedef float    v8f  __attribute__((ext_vector_type(8)));

// ---------------------------------------------------------------------------
// Phase 0: convert W2 [64, 9216] f32 -> f16 into workspace (1.1 MB, trivial)
// ---------------------------------------------------------------------------
__global__ __launch_bounds__(256)
void w2_to_f16_kernel(const float* __restrict__ W2, _Float16* __restrict__ W2h) {
    int i = blockIdx.x * blockDim.x + threadIdx.x;
    if (i < OUT_DIM * HIDDEN) W2h[i] = (_Float16)W2[i];
}

// ---------------------------------------------------------------------------
// Async helper: GLOBAL_LOAD_ASYNC_TO_LDS_B32 (ASYNCcnt-tracked, no VGPR stage).
// W1 is streamed exactly once -> non-temporal hint so it does not evict the
// freshly written H (75.5 MB) from the 192 MB L2 before phase 2 re-reads it.
// ---------------------------------------------------------------------------
__device__ __forceinline__ void async_copy_b32_nt(const float* gptr, float* lptr) {
    uint32_t lds_addr = (uint32_t)(uintptr_t)lptr;      // low 32 bits = LDS offset
    uint64_t g_addr   = (uint64_t)(uintptr_t)gptr;
    asm volatile("global_load_async_to_lds_b32 %0, %1, off th:TH_LOAD_NT"
                 :: "v"(lds_addr), "v"(g_addr) : "memory");
}

// Issue one W1 row (TOTAL floats) as exactly ASYNC_PER_ROW asyncs per thread
// (uniform per-wave ASYNCcnt; tail index clamped -> duplicate benign writes).
__device__ __forceinline__ void issue_row_async(const float* __restrict__ wrow,
                                                float* __restrict__ dst, int tid) {
#pragma unroll
    for (int it = 0; it < ASYNC_PER_ROW; ++it) {
        int i = tid + it * 1024;
        if (i > TOTAL - 1) i = TOTAL - 1;
        async_copy_b32_nt(wrow + i, dst + i);
    }
}

// ---------------------------------------------------------------------------
// Phase 1: H[b][j] = relu(row_j[a_b] + row_j[16384+g_b] + y_b*row_j[18432] + b1[j])
// One workgroup (1024 threads) owns JT=16 consecutive j's and ALL 4096 batches.
// Double-buffered LDS rows; row j+1 streams in via async-to-LDS while row j is
// gathered. W1 is read from HBM exactly once across the grid.
// ---------------------------------------------------------------------------
extern "C" __global__ __launch_bounds__(1024)
void encoder_hidden_kernel(const int*   __restrict__ artist,
                           const int*   __restrict__ genre,
                           const float* __restrict__ year,
                           const float* __restrict__ W1,
                           const float* __restrict__ b1,
                           _Float16*    __restrict__ H) {
    extern __shared__ float row[];            // 2 * ROWPITCH floats (147,520 B)
    const int tid = threadIdx.x;
    const int j0  = blockIdx.x * JT;
    const int b   = 4 * tid;                  // 4 consecutive batches per thread

    // per-thread batch metadata, vector-loaded once, lives in registers
    int4   ai = *(const int4*)(artist + b);
    int4   gi = *(const int4*)(genre  + b);
    float4 yn = *(const float4*)(year + b);
    int   a[4] = { ai.x, ai.y, ai.z, ai.w };
    int   g[4] = { gi.x + ARTIST_V, gi.y + ARTIST_V, gi.z + ARTIST_V, gi.w + ARTIST_V };
    float y[4] = { yn.x, yn.y, yn.z, yn.w };

    _Float16 acc[4][JT];                      // 64 halfs -> registers (loop fully unrolled)

    // Prologue: start streaming row j0 into buffer 0
    issue_row_async(W1 + (size_t)j0 * TOTAL, &row[0], tid);

#pragma unroll
    for (int jj = 0; jj < JT; ++jj) {
        float* cur = &row[(jj & 1) * ROWPITCH];

        if (jj + 1 < JT) {
            // Start next row into the other buffer (safe: barrier at end of the
            // previous iteration guaranteed all gathers from it completed).
            issue_row_async(W1 + (size_t)(j0 + jj + 1) * TOTAL,
                            &row[((jj + 1) & 1) * ROWPITCH], tid);
            // Retire everything except the ASYNC_PER_ROW just issued -> cur done.
            asm volatile("s_wait_asynccnt 0x13" ::: "memory");
        } else {
            asm volatile("s_wait_asynccnt 0x0" ::: "memory");
        }
        __syncthreads();                      // all waves' copies of cur complete

        const float yw = cur[TOTAL - 1];      // year coefficient (last column)
        const float bb = b1[j0 + jj];
#pragma unroll
        for (int l = 0; l < 4; ++l) {
            float h = cur[a[l]] + cur[g[l]] + y[l] * yw + bb;   // LDS gathers
            acc[l][jj] = (_Float16)fmaxf(h, 0.0f);              // relu, f16
        }
        __syncthreads();                      // protect cur before it is re-filled
    }

    // write H[b][j0..j0+15]: 32 contiguous bytes per batch, 16B-aligned
#pragma unroll
    for (int l = 0; l < 4; ++l) {
        _Float16* dst = H + (size_t)(b + l) * HIDDEN + j0;
        v8h v0, v1;
#pragma unroll
        for (int e = 0; e < 8; ++e) { v0[e] = acc[l][e]; v1[e] = acc[l][8 + e]; }
        *(v8h*)(dst)     = v0;
        *(v8h*)(dst + 8) = v1;
    }
}

// ---------------------------------------------------------------------------
// Phase 2: out[4096,64] = H[4096,9216](f16) @ W2h^T + b2 via WMMA f32_16x16x32_f16
// One wave per 16x32 (batch x out) macro-tile: a single A load feeds TWO WMMAs
// (two n-tiles, two independent accumulator chains) -> H traffic halved and the
// WMMA D->C dependency chain is broken for better pipelining. K-loop of 288.
// Per-lane layouts used here (A and B share contiguous K order):
//   A (16x32 f16): lanes 0-15 (M=lane) hold K 0..15; lanes 16-31 hold K 16..31 (contiguous)
//   B (32x16 f16): lane n&15 = column N; lanes>=16 hold K 16..31 (16 contiguous halfs)
//   D (16x16 f32): VGPR v -> M = v + 8*(lane>=16), N = lane&15
// ---------------------------------------------------------------------------
extern "C" __global__ __launch_bounds__(32)
void encoder_out_kernel(const _Float16* __restrict__ H,
                        const _Float16* __restrict__ W2h,
                        const float*    __restrict__ b2,
                        float*          __restrict__ out) {
    const int lane = threadIdx.x;
    const int b0 = blockIdx.x * 16;           // batch tile
    const int n0 = blockIdx.y * 32;           // out-dim macro-tile (2 x 16)
    const int mr = lane & 15;
    const int hg = lane >> 4;                 // half-group: 0 or 1

    const _Float16* Abase  = H   + (size_t)(b0 + mr) * HIDDEN + hg * 16;
    const _Float16* Bbase0 = W2h + (size_t)(n0 + mr) * HIDDEN + hg * 16;
    const _Float16* Bbase1 = Bbase0 + (size_t)16 * HIDDEN;

    v8f c0 = {0.f, 0.f, 0.f, 0.f, 0.f, 0.f, 0.f, 0.f};
    v8f c1 = {0.f, 0.f, 0.f, 0.f, 0.f, 0.f, 0.f, 0.f};

    for (int k = 0; k < HIDDEN; k += 32) {
        v8h alo = *(const v8h*)(Abase + k);        // K = hg*16 + 0..7    (16B aligned)
        v8h ahi = *(const v8h*)(Abase + k + 8);    // K = hg*16 + 8..15
        v16h a  = __builtin_shufflevector(alo, ahi,
                    0,1,2,3,4,5,6,7, 8,9,10,11,12,13,14,15);
        v16h bm0 = *(const v16h*)(Bbase0 + k);     // K = hg*16 + 0..15  (32B aligned)
        v16h bm1 = *(const v16h*)(Bbase1 + k);
        c0 = __builtin_amdgcn_wmma_f32_16x16x32_f16(
                false, a, false, bm0, (short)0, c0, false, false);
        c1 = __builtin_amdgcn_wmma_f32_16x16x32_f16(
                false, a, false, bm1, (short)0, c1, false, false);
    }

    const float bias0 = b2[n0 + mr];
    const float bias1 = b2[n0 + 16 + mr];
#pragma unroll
    for (int v = 0; v < 8; ++v) {
        const size_t r = (size_t)(b0 + v + hg * 8) * OUT_DIM;
        out[r + (n0 + mr)]      = c0[v] + bias0;
        out[r + (n0 + 16 + mr)] = c1[v] + bias1;
    }
}

// ---------------------------------------------------------------------------
// Launcher. Workspace layout: [ H f16 : 75,497,472 B ][ W2h f16 : 1,179,648 B ]
// ---------------------------------------------------------------------------
extern "C" void kernel_launch(void* const* d_in, const int* in_sizes, int n_in,
                              void* d_out, int out_size, void* d_ws, size_t ws_size,
                              hipStream_t stream) {
    const int*   artist = (const int*)  d_in[0];
    const int*   genre  = (const int*)  d_in[1];
    const float* year   = (const float*)d_in[2];
    const float* W1     = (const float*)d_in[3];
    const float* b1     = (const float*)d_in[4];
    const float* W2     = (const float*)d_in[5];
    const float* b2     = (const float*)d_in[6];
    float* out = (float*)d_out;

    _Float16* H   = (_Float16*)d_ws;
    _Float16* W2h = (_Float16*)((char*)d_ws + (size_t)BATCH * HIDDEN * sizeof(_Float16));

    // Phase 0: W2 -> f16
    w2_to_f16_kernel<<<(OUT_DIM * HIDDEN + 255) / 256, 256, 0, stream>>>(W2, W2h);

    // Phase 1: hidden layer (W1 read once; async global->LDS double buffering)
    const size_t lds_bytes = (size_t)2 * ROWPITCH * sizeof(float);  // 147,520 B < 320 KB/WGP
    encoder_hidden_kernel<<<HIDDEN / JT, 1024, lds_bytes, stream>>>(
        artist, genre, year, W1, b1, H);

    // Phase 2: WMMA GEMM -> out (one wave per 16x32 macro-tile)
    dim3 g2(BATCH / 16, OUT_DIM / 32);
    encoder_out_kernel<<<g2, 32, 0, stream>>>(H, W2h, b2, out);
}
